// CDMTransformer_78804059947147
// MI455X (gfx1250) — matrix operand order repaired and checked by reading, not compiled, not run on hardware
//
#include <hip/hip_runtime.h>
#include <hip/hip_bf16.h>

// ---------------------------------------------------------------------------
// CDNA5 (gfx1250) implementation of the 4-layer gamma-attention transformer.
// GEMMs: v_wmma_f32_16x16x32_bf16 with 32x32 register blocking per wave; the
// shared weight panel is staged into LDS by the Tensor Data Mover (TDM,
// tensor_load_to_lds + s_wait_tensorcnt), double-buffered.  Softmax/cumsum/
// layernorm stay in f32 VALU.  B=2, S=1024, D=1024, H=16, DH=64, L=4.
// ---------------------------------------------------------------------------

typedef __bf16  bf16_t;
typedef __attribute__((ext_vector_type(16))) __bf16 v16bf;
typedef __attribute__((ext_vector_type(8)))  __bf16 v8bf;
typedef __attribute__((ext_vector_type(8)))  float  v8f;
typedef __attribute__((ext_vector_type(4)))  unsigned int u32x4;
typedef __attribute__((ext_vector_type(8)))  int i32x8;
typedef __attribute__((ext_vector_type(4)))  int i32x4;

#define CB   2
#define CS   1024
#define CD   1024
#define CH   16
#define CDH  64
#define CL   4
#define CM   (CB * CS)          // 2048 token rows
#define CDD  (CD * CD)          // 1048576
#define NEGC (-1.0e32f)

// ---- WMMA wrapper ---------------------------------------------------------
static __device__ __forceinline__ v8f wmma_bf16(v16bf a, v16bf b, v8f c) {
  return __builtin_amdgcn_wmma_f32_16x16x32_bf16(
      /*neg_a=*/false, a, /*neg_b=*/false, b,
      /*c_mod=*/(short)0, c, /*reuse_a=*/false, /*reuse_b=*/false);
}

// A-matrix fragment (16x32 bf16, M=lane&15): lane-half h holds K runs
// [8h..8h+7] and [16+8h..16+8h+7].  Two contiguous 16B loads.
static __device__ __forceinline__ v16bf load_a_frag(const bf16_t* p, int hl) {
  const v8bf lo = *reinterpret_cast<const v8bf*>(p + 8 * hl);
  const v8bf hi = *reinterpret_cast<const v8bf*>(p + 8 * hl + 16);
  v16bf f;
#pragma unroll
  for (int i = 0; i < 8; ++i) { f[i] = lo[i]; f[i + 8] = hi[i]; }
  return f;
}

// B-matrix fragment (32x16 bf16, N=lane&15): lane-half h holds K runs
// [8h..8h+7] and [16+8h..16+8h+7].  Two contiguous 16B loads (global or LDS).
static __device__ __forceinline__ v16bf load_b_frag(const bf16_t* p, int hl) {
  const v8bf lo = *reinterpret_cast<const v8bf*>(p + 8 * hl);
  const v8bf hi = *reinterpret_cast<const v8bf*>(p + 8 * hl + 16);
  v16bf f;
#pragma unroll
  for (int i = 0; i < 8; ++i) { f[i] = lo[i]; f[i + 8] = hi[i]; }
  return f;
}

// ---- TDM: 2D tile (bf16) from global -> LDS per CDNA5 ISA ch8 D# layout ----
// tile_d0 elements contiguous per line, tile_d1 lines, line stride d0_stride
// elements; packed contiguously into LDS at lds_byte.
static __device__ __forceinline__ void tdm_load_2d(
    unsigned int lds_byte, const void* gptr,
    unsigned int tensor_d0, unsigned int tensor_d1,
    unsigned int tile_d0, unsigned int tile_d1,
    unsigned long long d0_stride /* elements */) {
  const unsigned long long ga = (unsigned long long)(size_t)gptr;
  u32x4 g0;
  g0[0] = 1u;                                        // count=1 (valid), user
  g0[1] = lds_byte;                                  // lds_addr
  g0[2] = (unsigned int)(ga & 0xffffffffu);          // global_addr[31:0]
  g0[3] = (unsigned int)((ga >> 32) & 0x01ffffffu) | (2u << 30);  // type=2
  i32x8 g1;
  g1[0] = (int)(1u << 16);                           // data_size=1 (2 bytes)
  g1[1] = (int)((tensor_d0 & 0xffffu) << 16);        // abar=0 | tdim0[15:0]
  g1[2] = (int)((tensor_d0 >> 16) | ((tensor_d1 & 0xffffu) << 16));
  g1[3] = (int)((tensor_d1 >> 16) | (tile_d0 << 16));
  g1[4] = (int)(tile_d1 & 0xffffu);                  // tile_dim1 | tile_dim2=0
  g1[5] = (int)(unsigned int)(d0_stride & 0xffffffffu);
  g1[6] = (int)(unsigned int)((d0_stride >> 32) & 0xffffu);  // dim1_stride=0
  g1[7] = 0;
  const i32x4 z4 = {0, 0, 0, 0};
#if __clang_major__ >= 23
  const i32x8 z8 = {0, 0, 0, 0, 0, 0, 0, 0};
  __builtin_amdgcn_tensor_load_to_lds(g0, g1, z4, z4, z8, 0);
#else
  __builtin_amdgcn_tensor_load_to_lds(g0, g1, z4, z4, 0);
#endif
}

// ---- converters -----------------------------------------------------------
__global__ __launch_bounds__(256) void cdm_cvt_bf16(const float* __restrict__ in,
                                                    bf16_t* __restrict__ out, int n) {
  int i = blockIdx.x * 256 + threadIdx.x;
  if (i < n) out[i] = (bf16_t)in[i];
}

// Transpose each DxD slab of an (L,D,D) f32 tensor into bf16 WT[l][n*D+k].
__global__ __launch_bounds__(256) void cdm_cvt_wT(const float* __restrict__ in,
                                                  bf16_t* __restrict__ out) {
  int i = blockIdx.x * 256 + threadIdx.x;      // i over L*D*D
  int l   = i >> 20;
  int rem = i & (CDD - 1);
  int k   = rem >> 10;
  int n   = rem & (CD - 1);
  out[l * CDD + n * CD + k] = (bf16_t)in[i];
}

// ---- GEMM: C[M=2048,N=1024] = A(bf16 row-major) @ WT(bf16, n-major) + bias -
// grid = (N/128, M/64); block = 256 (8 waves, 2x4); wave tile = 32x32.
// Weight panel (128 N-rows x 32 K) staged into LDS via TDM, double-buffered.
__global__ __launch_bounds__(256) void cdm_gemm_bias(
    const bf16_t* __restrict__ A, const bf16_t* __restrict__ WT,
    const float* __restrict__ bias,
    float* __restrict__ outF, bf16_t* __restrict__ outB, bf16_t* __restrict__ outVT) {
  __shared__ __align__(16) bf16_t bufB[2][128 * 32];   // 2 x 8 KB

  const int tid  = threadIdx.x;
  const int wave = tid >> 5, lane = tid & 31;
  const int r = lane & 15, hl = lane >> 4;
  const int rowb   = blockIdx.y * 64 + (wave >> 2) * 32;
  const int blkcol = blockIdx.x * 128;
  const int wc     = (wave & 3) * 32;                  // wave col within block

  const bf16_t* Ap0 = A + (size_t)(rowb + r)      * CD;   // K contiguous
  const bf16_t* Ap1 = A + (size_t)(rowb + 16 + r) * CD;
  const bf16_t* Wpanel = WT + (size_t)blkcol * CD;        // block's B panel

  const unsigned int ldsB[2] = {
      (unsigned int)(size_t)(void*)&bufB[0][0],
      (unsigned int)(size_t)(void*)&bufB[1][0]};

  // preload K-chunk 0 (TDM issued by wave 0; EXEC ignored by tensor ops)
  if (wave == 0) {
    tdm_load_2d(ldsB[0], Wpanel, CD, CD, 32, 128, CD);
    __builtin_amdgcn_s_wait_tensorcnt(0);
  }
  __syncthreads();

  v8f acc00 = {}, acc01 = {}, acc10 = {}, acc11 = {};
#pragma unroll 2
  for (int k0 = 0; k0 < CD; k0 += 32) {
    const int cur = (k0 >> 5) & 1;
    const bool more = (k0 + 32) < CD;
    if (wave == 0 && more)
      tdm_load_2d(ldsB[cur ^ 1], Wpanel + (k0 + 32), CD, CD, 32, 128, CD);

    if ((k0 & 63) == 0) {            // lane row line = 128B -> every 64 K
      __builtin_prefetch(Ap0 + k0 + 64, 0, 1);
      __builtin_prefetch(Ap1 + k0 + 64, 0, 1);
    }
    v16bf a0 = load_a_frag(Ap0 + k0, hl);
    v16bf a1 = load_a_frag(Ap1 + k0, hl);
    const bf16_t* Lb = &bufB[cur][(wc + r) * 32];        // LDS-resident panel
    v16bf b0 = load_b_frag(Lb, hl);
    v16bf b1 = load_b_frag(Lb + 16 * 32, hl);
    acc00 = wmma_bf16(a0, b0, acc00);
    acc01 = wmma_bf16(a0, b1, acc01);
    acc10 = wmma_bf16(a1, b0, acc10);
    acc11 = wmma_bf16(a1, b1, acc11);

    if (wave == 0 && more) __builtin_amdgcn_s_wait_tensorcnt(0);
    __syncthreads();
  }

  const float bv0 = bias[blkcol + wc + r];
  const float bv1 = bias[blkcol + wc + 16 + r];
#pragma unroll
  for (int sub = 0; sub < 4; ++sub) {
    const v8f&  acc  = (sub == 0) ? acc00 : (sub == 1) ? acc01
                      : (sub == 2) ? acc10 : acc11;
    const int   rofs = (sub >> 1) * 16;
    const int   cofs = (sub & 1) * 16;
    const int   colg = blkcol + wc + cofs + r;
    const float bv   = (sub & 1) ? bv1 : bv0;
#pragma unroll
    for (int i = 0; i < 8; ++i) {
      const int rowg = rowb + rofs + i + 8 * hl;
      const float v  = acc[i] + bv;
      if (outF)  outF[(size_t)rowg * CD + colg] = v;
      if (outB)  outB[(size_t)rowg * CD + colg] = (bf16_t)v;
      if (outVT) {                                   // Vt[b][d][s]
        const int bb = rowg >> 10, ss = rowg & (CS - 1);
        outVT[(size_t)bb * CDD + (size_t)colg * CS + ss] = (bf16_t)v;
      }
    }
  }
}

// ---- attention: one block per (b, h, 16-query tile); 256 threads (8 waves) -
__global__ __launch_bounds__(256) void cdm_attention(
    const bf16_t* __restrict__ Qb,    // (B*S, D) bf16, shared q/k projection
    const bf16_t* __restrict__ Vt,    // (B, D, S) bf16
    const float*  __restrict__ gamma_l,  // H floats for this layer
    bf16_t* __restrict__ Ob) {        // (B*S, D) bf16 attention output
  __shared__ float sc[16 * CS];       // 64 KB score/attn rows
  __shared__ float red[16 * 17];
  __shared__ float part[4 * 16 * 16];

  const int tid  = threadIdx.x;
  const int wave = tid >> 5, lane = tid & 31;
  const int r = lane & 15, hl = lane >> 4;

  const int bi = blockIdx.x >> 10;
  const int h  = (blockIdx.x >> 6) & (CH - 1);
  const int qt = blockIdx.x & 63;
  const int qb = qt * 16;

  // ---- phase 1: scores = (Q Q^T) / sqrt(DH) via WMMA ----
  const bf16_t* Qrow = Qb + (size_t)(bi * CS + qb + r) * CD + h * CDH;
  const v16bf a0 = load_a_frag(Qrow +  0, hl);
  const v16bf a1 = load_a_frag(Qrow + 32, hl);
#pragma unroll
  for (int t = 0; t < 8; ++t) {
    const int nt = wave * 8 + t;
    const int kb = nt * 16;
    const bf16_t* Krow = Qb + (size_t)(bi * CS + kb + r) * CD + h * CDH;
    v16bf b0 = load_b_frag(Krow +  0, hl);
    v16bf b1 = load_b_frag(Krow + 32, hl);
    v8f acc = {};
    acc = wmma_bf16(a0, b0, acc);
    acc = wmma_bf16(a1, b1, acc);
#pragma unroll
    for (int i = 0; i < 8; ++i)
      sc[(i + 8 * hl) * CS + kb + r] = acc[i] * 0.125f;
  }
  __syncthreads();

  // ---- phase 2: softmax -> cumsum decay -> softmax -> maxout (f32 VALU) ----
  {
    const int row = tid >> 4, seg = tid & 15;
    const int c0 = seg * 64;
    const int qg = qb + row;
    float* srow = &sc[row * CS];

    float mx = -3.0e38f;
    for (int j = 0; j < 64; ++j) {
      const int c = c0 + j;
      const float mv = (qg > c) ? srow[c] : NEGC;
      mx = fmaxf(mx, mv);
    }
    red[row * 17 + seg] = mx;
    __syncthreads();
    float rmx = -3.0e38f;
    for (int j = 0; j < 16; ++j) rmx = fmaxf(rmx, red[row * 17 + j]);
    __syncthreads();

    float se = 0.0f;
    for (int j = 0; j < 64; ++j) {
      const int c = c0 + j;
      const float mv = (qg > c) ? srow[c] : NEGC;
      se += __expf(mv - rmx);
    }
    red[row * 17 + seg] = se;
    __syncthreads();
    float rsum = 0.0f, pre = 0.0f;
    for (int j = 0; j < 16; ++j) {
      const float t = red[row * 17 + j];
      rsum += t;
      if (j < seg) pre += t;
    }
    __syncthreads();

    // cumsum of p; disttotal == 1 exactly (softmax).  Apply decay effect.
    const float inv = 1.0f / rsum;
    float run = pre * inv;
    const float g = -fabsf(gamma_l[h]);
    for (int j = 0; j < 64; ++j) {
      const int c = c0 + j;
      const float sv = srow[c];
      const float mv = (qg > c) ? sv : NEGC;
      run += __expf(mv - rmx) * inv;
      float dd = (1.0f - run) * fabsf((float)(qg - c));
      dd = fmaxf(dd, 0.0f);
      float eff = __expf(__builtin_sqrtf(dd) * g);
      eff = fminf(fmaxf(eff, 1e-5f), 1e5f);
      srow[c] = sv * eff;                 // s2 = scores * effect
    }
    __syncthreads();

    // second softmax over masked s2
    float mx2 = -3.0e38f;
    for (int j = 0; j < 64; ++j) {
      const int c = c0 + j;
      const float mv = (qg > c) ? srow[c] : NEGC;
      mx2 = fmaxf(mx2, mv);
    }
    red[row * 17 + seg] = mx2;
    __syncthreads();
    float rmx2 = -3.0e38f;
    for (int j = 0; j < 16; ++j) rmx2 = fmaxf(rmx2, red[row * 17 + j]);
    __syncthreads();
    float se2 = 0.0f;
    for (int j = 0; j < 64; ++j) {
      const int c = c0 + j;
      const float mv = (qg > c) ? srow[c] : NEGC;
      se2 += __expf(mv - rmx2);
    }
    red[row * 17 + seg] = se2;
    __syncthreads();
    float rsum2 = 0.0f;
    for (int j = 0; j < 16; ++j) rsum2 += red[row * 17 + j];
    __syncthreads();

    // maxout: attn.max == 1/rsum2 -> scale = min(rsum2, 5)
    const float factor = fminf(rsum2, 5.0f) / rsum2;
    for (int j = 0; j < 64; ++j) {
      const int c = c0 + j;
      const float s2 = srow[c];
      srow[c] = (qg > c) ? __expf(s2 - rmx2) * factor : 0.0f;
    }
  }
  __syncthreads();

  // ---- phase 3: out(16x64) = attn(16x1024) @ V(1024x64) via WMMA ----
  {
    const int nt = wave & 3;       // which 16-wide slice of DH
    const int kh = wave >> 2;      // K halves split across wave pairs
    const bf16_t* Vcol =
        Vt + (size_t)bi * CDD + (size_t)(h * CDH + nt * 16 + r) * CS;
    v8f acc = {};
#pragma unroll 4
    for (int ks = 0; ks < 16; ++ks) {
      const int k0 = kh * 512 + ks * 32;
      // A fragment from LDS f32 attn -> bf16 (split-run layout)
      const float* ap = &sc[r * CS + k0 + 8 * hl];
      v16bf a;
#pragma unroll
      for (int i = 0; i < 8; ++i) {
        a[i]     = (bf16_t)ap[i];
        a[i + 8] = (bf16_t)ap[i + 16];
      }
      v16bf b = load_b_frag(Vcol + k0, hl);
      acc = wmma_bf16(a, b, acc);
    }
    if (kh == 1) {
#pragma unroll
      for (int i = 0; i < 8; ++i)
        part[nt * 256 + (i + 8 * hl) * 16 + r] = acc[i];
    }
    __syncthreads();
    if (kh == 0) {
#pragma unroll
      for (int i = 0; i < 8; ++i) {
        const float v = acc[i] + part[nt * 256 + (i + 8 * hl) * 16 + r];
        Ob[(size_t)(bi * CS + qb + i + 8 * hl) * CD + h * CDH + nt * 16 + r] =
            (bf16_t)v;
      }
    }
  }
}

// ---- residual + LayerNorm (one block per token row) -----------------------
__global__ __launch_bounds__(256) void cdm_layernorm(
    const float* __restrict__ X, const float* __restrict__ Aadd,
    const float* __restrict__ gam, const float* __restrict__ bet,
    float* __restrict__ outF, bf16_t* __restrict__ outB) {
  __shared__ float rb[256];
  const int tid = threadIdx.x;
  const size_t base = (size_t)blockIdx.x * CD;

  float x[4];
  float s = 0.0f;
#pragma unroll
  for (int i = 0; i < 4; ++i) {
    const int c = tid + 256 * i;
    float v = X[base + c];
    if (Aadd) v += Aadd[base + c];
    x[i] = v;
    s += v;
  }
  rb[tid] = s;
  __syncthreads();
  for (int off = 128; off > 0; off >>= 1) {
    if (tid < off) rb[tid] += rb[tid + off];
    __syncthreads();
  }
  const float mu = rb[0] * (1.0f / CD);
  __syncthreads();

  float s2 = 0.0f;
#pragma unroll
  for (int i = 0; i < 4; ++i) { const float d = x[i] - mu; s2 += d * d; }
  rb[tid] = s2;
  __syncthreads();
  for (int off = 128; off > 0; off >>= 1) {
    if (tid < off) rb[tid] += rb[tid + off];
    __syncthreads();
  }
  const float rstd = rsqrtf(rb[0] * (1.0f / CD) + 1e-5f);

#pragma unroll
  for (int i = 0; i < 4; ++i) {
    const int c = tid + 256 * i;
    const float y = (x[i] - mu) * rstd * gam[c] + bet[c];
    outF[base + c] = y;
    if (outB) outB[base + c] = (bf16_t)y;
  }
}

// ---------------------------------------------------------------------------
extern "C" void kernel_launch(void* const* d_in, const int* in_sizes, int n_in,
                              void* d_out, int out_size, void* d_ws, size_t ws_size,
                              hipStream_t stream) {
  (void)in_sizes; (void)n_in; (void)out_size; (void)ws_size;
  const float* q      = (const float*)d_in[0];
  // d_in[1] = lens (unused in eval mode)
  const float* Wq     = (const float*)d_in[2];
  const float* bq     = (const float*)d_in[3];
  const float* Wv     = (const float*)d_in[4];
  const float* bv     = (const float*)d_in[5];
  const float* Wo     = (const float*)d_in[6];
  const float* bo     = (const float*)d_in[7];
  const float* gammas = (const float*)d_in[8];
  const float* ln_g   = (const float*)d_in[9];
  const float* ln_b   = (const float*)d_in[10];
  const float* fin_g  = (const float*)d_in[11];
  const float* fin_b  = (const float*)d_in[12];

  char*  ws  = (char*)d_ws;
  size_t off = 0;
  auto alloc = [&](size_t bytes) -> void* {
    void* p = ws + off;
    off += (bytes + 255) & ~(size_t)255;
    return p;
  };
  bf16_t* Xb  = (bf16_t*)alloc((size_t)CM * CD * 2);        // activations bf16
  float*  Xf  = (float*) alloc((size_t)CM * CD * 4);        // activations f32
  bf16_t* WqT = (bf16_t*)alloc((size_t)CL * CDD * 2);
  bf16_t* WvT = (bf16_t*)alloc((size_t)CL * CDD * 2);
  bf16_t* WoT = (bf16_t*)alloc((size_t)CL * CDD * 2);
  bf16_t* Qp  = (bf16_t*)alloc((size_t)CM * CD * 2);        // q/k projection
  bf16_t* Vt  = (bf16_t*)alloc((size_t)CB * CDD * 2);       // V transposed [b][d][s]
  bf16_t* Ob  = (bf16_t*)alloc((size_t)CM * CD * 2);        // attention out
  float*  AO  = (float*) alloc((size_t)CM * CD * 4);        // O-proj out f32

  // one-time weight transpose+convert (all layers)
  {
    const int n = CL * CDD;
    cdm_cvt_wT<<<n / 256, 256, 0, stream>>>(Wq, WqT);
    cdm_cvt_wT<<<n / 256, 256, 0, stream>>>(Wv, WvT);
    cdm_cvt_wT<<<n / 256, 256, 0, stream>>>(Wo, WoT);
  }
  // initial activation -> bf16
  cdm_cvt_bf16<<<(CM * CD) / 256, 256, 0, stream>>>(q, Xb, CM * CD);

  const dim3 ggrid(CD / 128, CM / 64);
  for (int l = 0; l < CL; ++l) {
    // Q/K projection (shared) and V projection (stored transposed)
    cdm_gemm_bias<<<ggrid, 256, 0, stream>>>(Xb, WqT + (size_t)l * CDD,
                                             bq + l * CD, nullptr, Qp, nullptr);
    cdm_gemm_bias<<<ggrid, 256, 0, stream>>>(Xb, WvT + (size_t)l * CDD,
                                             bv + l * CD, nullptr, nullptr, Vt);
    // gamma attention
    cdm_attention<<<CB * CH * (CS / 16), 256, 0, stream>>>(Qp, Vt,
                                                           gammas + l * CH, Ob);
    // output projection
    cdm_gemm_bias<<<ggrid, 256, 0, stream>>>(Ob, WoT + (size_t)l * CDD,
                                             bo + l * CD, AO, nullptr, nullptr);
    // residual + layernorm -> next layer activations (f32 + bf16)
    cdm_layernorm<<<CM, 256, 0, stream>>>((l == 0) ? q : Xf, AO,
                                          ln_g + l * CD, ln_b + l * CD, Xf, Xb);
  }
  // final layernorm -> output (f32)
  cdm_layernorm<<<CM, 256, 0, stream>>>(Xf, nullptr, fin_g, fin_b,
                                        (float*)d_out, nullptr);
}
